// KNN_81312320848146
// MI455X (gfx1250) — compile-verified, matrix operand order checked by reading, not run on hardware
//
#include <hip/hip_runtime.h>

typedef __attribute__((ext_vector_type(16))) __bf16        bf16x16;
typedef __attribute__((ext_vector_type(8)))  float         floatx8;
typedef __attribute__((ext_vector_type(4)))  unsigned int  uint32x4;
typedef __attribute__((ext_vector_type(8)))  int           int32x8;
typedef __attribute__((ext_vector_type(4)))  int           int32x4;

#define B_Q   2048
#define N_T   50000
#define F_D   256
#define O_D   24
#define NSPLIT 25
#define CHUNK  2000            // N_T / NSPLIT, divisible by 16
#define NTILES (CHUNK / 16)    // 125
#define KSTEPS (F_D / 32)      // 8

// ws layout (bytes):
//   Xbf  @ 0          : 50000 rows x 1024 B  (hi 512B | lo 512B per row) = 51,200,000
//   t2   @ 51,200,000 : 50000 f32
//   pval @ 51,404,800 : 128*25*16 f32
//   pidx @ 51,609,600 : 128*25*16 i32
#define WS_T2   51200000u
#define WS_PVAL 51404800u
#define WS_PIDX 51609600u

__device__ __forceinline__ void splitpack(float v0, float v1,
                                          unsigned int& hi, unsigned int& lo) {
  __bf16 h0 = (__bf16)v0, h1 = (__bf16)v1;
  __bf16 l0 = (__bf16)(v0 - (float)h0), l1 = (__bf16)(v1 - (float)h1);
  unsigned short uh0 = __builtin_bit_cast(unsigned short, h0);
  unsigned short uh1 = __builtin_bit_cast(unsigned short, h1);
  unsigned short ul0 = __builtin_bit_cast(unsigned short, l0);
  unsigned short ul1 = __builtin_bit_cast(unsigned short, l1);
  hi = (unsigned int)uh0 | ((unsigned int)uh1 << 16);
  lo = (unsigned int)ul0 | ((unsigned int)ul1 << 16);
}

// TDM: 2D tile (nrows x 1024B, row stride 1024B) global -> LDS.
// Issued per-wave (EXEC ignored by TENSOR_LOAD_TO_LDS); tracked on this
// wave's TENSORcnt.
__device__ __forceinline__ void tdm_load_rows(unsigned int ldsOff,
                                              unsigned long long gaddr,
                                              unsigned int nrows) {
  uint32x4 g0;
  g0[0] = 0x1u;                                                // count=1, user mode
  g0[1] = ldsOff;                                              // lds_addr (bytes)
  g0[2] = (unsigned int)gaddr;                                 // global_addr[31:0]
  g0[3] = (unsigned int)((gaddr >> 32) & 0x01FFFFFFull)        // global_addr[56:32]
        | (2u << 30);                                          // type=2 (image)
  int32x8 g1;
  g1[0] = 0x00010000;          // workgroup_mask=0, data_size=1 (2 bytes/elem)
  g1[1] = (int)(512u << 16);   // tensor_dim0 = 512 elems (low16 in [31:16])
  g1[2] = 0;                   // tensor_dim0 hi16 = 0, tensor_dim1 lo16 = 0
  g1[3] = (int)((512u << 16) | 1u); // tensor_dim1 = 0x10000 rows; tile_dim0 = 512
  g1[4] = (int)nrows;          // tile_dim1 = nrows, tile_dim2 = 0 (2D)
  g1[5] = 512;                 // tensor_dim0_stride = 512 elems (1024 B)
  g1[6] = 0;
  g1[7] = 0;
  int32x4 z4 = {0, 0, 0, 0};
  int32x8 z8 = {0, 0, 0, 0, 0, 0, 0, 0};
  __builtin_amdgcn_tensor_load_to_lds(g0, g1, z4, z4, z8, 0);
}

// ---- Kernel 1: fused convert (f32 -> bf16 hi/lo planes) + row norms ----
__global__ void knn_prep(const float* __restrict__ Xt,
                         unsigned int* __restrict__ XbfU,   // [N][256] uints
                         float* __restrict__ t2) {
  int lane = threadIdx.x & 31;
  int wid  = threadIdx.x >> 5;
  int row  = blockIdx.x * 8 + wid;
  if (row >= N_T) return;
  const float4* p = (const float4*)(Xt + (size_t)row * F_D);
  float4 a = p[lane * 2];
  float4 b = p[lane * 2 + 1];
  float s = a.x*a.x + a.y*a.y + a.z*a.z + a.w*a.w
          + b.x*b.x + b.y*b.y + b.z*b.z + b.w*b.w;
  unsigned int h0,l0,h1,l1,h2,l2,h3,l3;
  splitpack(a.x, a.y, h0, l0);
  splitpack(a.z, a.w, h1, l1);
  splitpack(b.x, b.y, h2, l2);
  splitpack(b.z, b.w, h3, l3);
  unsigned int* rowp = XbfU + (size_t)row * 256;
  *(uint4*)(rowp + lane * 4)       = make_uint4(h0, h1, h2, h3);   // hi plane
  *(uint4*)(rowp + 128 + lane * 4) = make_uint4(l0, l1, l2, l3);   // lo plane
  #pragma unroll
  for (int m = 16; m >= 1; m >>= 1) s += __shfl_xor(s, m, 32);
  if (lane == 0) t2[row] = s;
}

// ---- Kernel 2: WMMA distance + partial argmin, TDM double-buffered B tiles ----
// grid: (32, NSPLIT); block: 128 threads = 4 waves (one 16-query tile each).
// Every wave DMAs its own 4-row slice of the 16-row B tile (symmetric, no
// divergent guards around the TDM op) and waits on its own TENSORcnt.
__launch_bounds__(128, 1)
__global__ void knn_gemm(const float* __restrict__ x,
                         const unsigned int* __restrict__ XbfU,
                         const float* __restrict__ t2,
                         float* __restrict__ pval,
                         int*   __restrict__ pidx) {
  __shared__ __align__(16) unsigned int lds[2][16 * 256];  // 2 x 16KB tiles

  const int lane  = threadIdx.x & 31;
  const int wid   = threadIdx.x >> 5;
  const int half  = lane >> 4;
  const int r     = lane & 15;
  const int btile = blockIdx.x * 4 + wid;   // 0..127
  const int chunk = blockIdx.y;             // 0..NSPLIT-1
  const int nbase = chunk * CHUNK;

  // Per-wave DMA slice: 4 rows x 1024 B.
  const unsigned long long xbase = (unsigned long long)(size_t)XbfU
                                 + (unsigned long long)nbase * 1024ull
                                 + (unsigned long long)(wid * 4096);
  const unsigned int ldsOff0 = (unsigned int)(size_t)&lds[0][0] + wid * 4096u;
  const unsigned int ldsOff1 = (unsigned int)(size_t)&lds[1][0] + wid * 4096u;

  // ---- A fragments (queries): bf16 hi/lo split, resident in VGPRs ----
  unsigned int ahi[KSTEPS][8], alo[KSTEPS][8];
  {
    const float* xrow = x + (size_t)(btile * 16 + r) * F_D;
    #pragma unroll
    for (int s = 0; s < KSTEPS; ++s) {
      #pragma unroll
      for (int i = 0; i < 8; ++i) {
        int k = s * 32 + ((i < 4) ? (8 * half + 2 * i) : (16 + 8 * half + 2 * (i - 4)));
        float2 v = *(const float2*)(xrow + k);
        splitpack(v.x, v.y, ahi[s][i], alo[s][i]);
      }
    }
  }

  float mval[8];
  int   midx[8];
  #pragma unroll
  for (int j = 0; j < 8; ++j) { mval[j] = 3.4e38f; midx[j] = 0; }

  // Prologue: every wave kicks off its slice of tile 0.
  tdm_load_rows(ldsOff0, xbase, 4u);

  for (int t = 0; t < NTILES; ++t) {
    const unsigned int* buf = lds[t & 1];
    __builtin_amdgcn_s_wait_tensorcnt(0);      // this wave's slice of tile t done
    __syncthreads();                           // whole tile t visible; prev reads done
    if ((t + 1) < NTILES) {
      tdm_load_rows((t & 1) ? ldsOff0 : ldsOff1,
                    xbase + (unsigned long long)(t + 1) * 16384ull, 4u);
    }

    floatx8 c = {0.f, 0.f, 0.f, 0.f, 0.f, 0.f, 0.f, 0.f};
    #pragma unroll
    for (int s = 0; s < KSTEPS; ++s) {
      int base0 = r * 256 + s * 16 + 4 * half;   // uint index (row = 256 uints)
      uint4 bh0 = *(const uint4*)&buf[base0];
      uint4 bh1 = *(const uint4*)&buf[base0 + 8];
      uint4 bl0 = *(const uint4*)&buf[base0 + 128];
      uint4 bl1 = *(const uint4*)&buf[base0 + 136];
      union { unsigned int u[8]; bf16x16 v; } bh, bl, ah, al;
      bh.u[0]=bh0.x; bh.u[1]=bh0.y; bh.u[2]=bh0.z; bh.u[3]=bh0.w;
      bh.u[4]=bh1.x; bh.u[5]=bh1.y; bh.u[6]=bh1.z; bh.u[7]=bh1.w;
      bl.u[0]=bl0.x; bl.u[1]=bl0.y; bl.u[2]=bl0.z; bl.u[3]=bl0.w;
      bl.u[4]=bl1.x; bl.u[5]=bl1.y; bl.u[6]=bl1.z; bl.u[7]=bl1.w;
      #pragma unroll
      for (int i = 0; i < 8; ++i) { ah.u[i] = ahi[s][i]; al.u[i] = alo[s][i]; }
      c = __builtin_amdgcn_wmma_f32_16x16x32_bf16(false, ah.v, false, bh.v, (short)0, c, false, false);
      c = __builtin_amdgcn_wmma_f32_16x16x32_bf16(false, ah.v, false, bl.v, (short)0, c, false, false);
      c = __builtin_amdgcn_wmma_f32_16x16x32_bf16(false, al.v, false, bh.v, (short)0, c, false, false);
    }

    int   nglob = nbase + t * 16 + r;          // N = lane&15 (C/D layout)
    float t2v   = t2[nglob];
    #pragma unroll
    for (int j = 0; j < 8; ++j) {              // C vgpr j -> query M = j + 8*half
      float score = t2v - 2.0f * c[j];
      if (score < mval[j]) { mval[j] = score; midx[j] = nglob; }
    }
  }

  // ---- argmin across the 16 lanes that share each query row ----
  #pragma unroll
  for (int j = 0; j < 8; ++j) {
    float v  = mval[j];
    int   ix = midx[j];
    #pragma unroll
    for (int m = 8; m >= 1; m >>= 1) {
      float ov  = __shfl_xor(v,  m, 32);
      int   oix = __shfl_xor(ix, m, 32);
      if (ov < v || (ov == v && oix < ix)) { v = ov; ix = oix; }
    }
    if (r == 0) {
      int M = j + 8 * half;
      int o = (btile * NSPLIT + chunk) * 16 + M;
      pval[o] = v;
      pidx[o] = ix;
    }
  }
}

// ---- Kernel 3: final argmin over chunks + gather Y ----
__global__ void knn_gather(const float* __restrict__ Yt,
                           const float* __restrict__ pval,
                           const int*   __restrict__ pidx,
                           float* __restrict__ out) {
  int b = blockIdx.x * blockDim.x + threadIdx.x;
  if (b >= B_Q) return;
  int btile = b >> 4, row = b & 15;
  float best = 3.4e38f;
  int   bi   = 0;
  for (int ck = 0; ck < NSPLIT; ++ck) {
    int o = (btile * NSPLIT + ck) * 16 + row;
    float v  = pval[o];
    int   ix = pidx[o];
    if (v < best || (v == best && ix < bi)) { best = v; bi = ix; }
  }
  const float* src = Yt + (size_t)bi * O_D;
  float* dst = out + (size_t)b * O_D;
  #pragma unroll
  for (int o = 0; o < O_D; ++o) dst[o] = src[o];
}

extern "C" void kernel_launch(void* const* d_in, const int* in_sizes, int n_in,
                              void* d_out, int out_size, void* d_ws, size_t ws_size,
                              hipStream_t stream) {
  const float* x  = (const float*)d_in[0];   // [2048, 256]
  const float* Xt = (const float*)d_in[1];   // [50000, 256]
  const float* Yt = (const float*)d_in[2];   // [50000, 24]
  float* out = (float*)d_out;                // [2048, 24]

  unsigned int* Xbf = (unsigned int*)d_ws;
  float* t2   = (float*)((char*)d_ws + WS_T2);
  float* pval = (float*)((char*)d_ws + WS_PVAL);
  int*   pidx = (int*)  ((char*)d_ws + WS_PIDX);

  knn_prep<<<dim3((N_T + 7) / 8), dim3(256), 0, stream>>>(Xt, Xbf, t2);
  knn_gemm<<<dim3(32, NSPLIT), dim3(128), 0, stream>>>(x, Xbf, t2, pval, pidx);
  knn_gather<<<dim3((B_Q + 255) / 256), dim3(256), 0, stream>>>(Yt, pval, pidx, out);
}
